// TreeHyperLISTA_18923625906628
// MI455X (gfx1250) — matrix operand order repaired and checked by reading, not compiled, not run on hardware
//
#include <hip/hip_runtime.h>
#include <hip/hip_bf16.h>
#include <math.h>

// ---------------- types for CDNA5 WMMA ----------------
typedef __bf16 bf16_t;
typedef bf16_t v16bf __attribute__((ext_vector_type(16)));
typedef bf16_t v8bf  __attribute__((ext_vector_type(8)));
typedef float  v8f   __attribute__((ext_vector_type(8)));

#define B_      32
#define M_      256
#define N_      32768
#define LAYERS_ 16
#define RHO_    0.5f
#define CHUNK_  512   // K-split chunk for the N-reduction GEMM

// ---------------- fragment loaders ----------------
// 16-bit A-matrix 16x32 (MxK), wave32:
//   lane l: m = l&15, half = l>>4
//   halves 0..7  of v16bf = K[k0 + half*8 .. +8)
//   halves 8..15 of v16bf = K[k0 + 16 + half*8 .. +8)
__device__ __forceinline__ v16bf load_afrag(const bf16_t* __restrict__ row,
                                            int k0, int half) {
  v8bf lo = *(const v8bf*)(row + k0 + half * 8);
  v8bf hi = *(const v8bf*)(row + k0 + 16 + half * 8);
  return __builtin_shufflevector(lo, hi, 0, 1, 2, 3, 4, 5, 6, 7,
                                         8, 9, 10, 11, 12, 13, 14, 15);
}
// 16-bit B-matrix 32x16 (KxN): lane l holds column n = l&15,
// K = [k0 + half*8 .. +8) then [k0 + 16 + half*8 .. +8) (same order as A).
__device__ __forceinline__ v16bf load_bfrag(const bf16_t* __restrict__ col,
                                            int k0, int half) {
  v8bf lo = *(const v8bf*)(col + k0 + half * 8);
  v8bf hi = *(const v8bf*)(col + k0 + 16 + half * 8);
  return __builtin_shufflevector(lo, hi, 0, 1, 2, 3, 4, 5, 6, 7,
                                         8, 9, 10, 11, 12, 13, 14, 15);
}

__device__ __forceinline__ v8f wmma_bf16(v16bf a, v16bf b, v8f c) {
  return __builtin_amdgcn_wmma_f32_16x16x32_bf16(false, a, false, b,
                                                 (short)0, c, false, false);
}

// ---------------- utility kernels ----------------
__global__ __launch_bounds__(256) void k_zero_f32(float* __restrict__ p, long n) {
  long i = (long)blockIdx.x * 256 + threadIdx.x;
  if (i < n) p[i] = 0.0f;
}

__global__ __launch_bounds__(256) void k_cvt_bf16(const float* __restrict__ s,
                                                  bf16_t* __restrict__ d, long n) {
  long i = (long)blockIdx.x * 256 + threadIdx.x;
  if (i < n) d[i] = (bf16_t)s[i];
}

// Wt[n*M + m] = (bf16) W[m*N + n]   (tiled 32x32 transpose)
__global__ __launch_bounds__(256) void k_transpose_cvt(const float* __restrict__ Wsrc,
                                                       bf16_t* __restrict__ Wt) {
  __shared__ float t[32][33];
  int n0 = blockIdx.x * 32, m0 = blockIdx.y * 32;
  for (int r = threadIdx.y; r < 32; r += 8)
    t[r][threadIdx.x] = Wsrc[(size_t)(m0 + r) * N_ + n0 + threadIdx.x];
  __syncthreads();
  for (int r = threadIdx.y; r < 32; r += 8)
    Wt[(size_t)(n0 + r) * M_ + m0 + threadIdx.x] = (bf16_t)t[threadIdx.x][r];
}

// ---------------- GEMM 1/3: acc[b,m] += sum_n x[b,n]*A[m,n] (K split) ----------------
// grid (16 mtiles, 2 btiles, 8), block 256 (8 waves; wave -> one K chunk)
// Manually unrolled x2 with dual accumulator chains so two WMMAs + two load
// groups are in flight per iteration.
__global__ __launch_bounds__(256)
void k_gemm_xAT(const bf16_t* __restrict__ xb, const bf16_t* __restrict__ Ab,
                float* __restrict__ acc) {
  const int lane = threadIdx.x & 31, wave = threadIdx.x >> 5;
  const int half = lane >> 4, q = lane & 15;
  const int mt = blockIdx.x, bt = blockIdx.y;
  const int ch = blockIdx.z * 8 + wave;                 // 0..63
  const bf16_t* xrow = xb + (size_t)(bt * 16 + q) * N_; // A-operand rows = b
  const bf16_t* bcol = Ab + (size_t)(mt * 16 + q) * N_; // B-operand cols = m
  v8f c0 = {}, c1 = {};
  const int k0b = ch * CHUNK_;
  for (int k0 = k0b; k0 < k0b + CHUNK_; k0 += 64) {
    // speculative prefetch of the stream 4KB ahead (global_prefetch_b8)
    __builtin_prefetch(bcol + k0 + 2048, 0, 3);
    v16bf a0 = load_afrag(xrow, k0, half);
    v16bf b0 = load_bfrag(bcol, k0, half);
    v16bf a1 = load_afrag(xrow, k0 + 32, half);
    v16bf b1 = load_bfrag(bcol, k0 + 32, half);
    c0 = wmma_bf16(a0, b0, c0);
    c1 = wmma_bf16(a1, b1, c1);
  }
  float* out = acc + (size_t)(bt * 16) * M_ + (mt * 16 + q);
#pragma unroll
  for (int v = 0; v < 8; ++v)
    atomicAdd(out + (size_t)(v + 8 * half) * M_, c0[v] + c1[v]);
}

// r[b,m] = y[b,m] - racc[b,m], stored as bf16 (only consumer needs bf16)
__global__ __launch_bounds__(256)
void k_residual(const float* __restrict__ y, const float* __restrict__ racc,
                bf16_t* __restrict__ rb) {
  int i = blockIdx.x * 256 + threadIdx.x;
  if (i < B_ * M_) rb[i] = (bf16_t)(y[i] - racc[i]);
}

// ---------------- GEMM 2: l1[b] += sum_n | sum_m r[b,m]*Apinv[n,m] | ----------------
// grid (256, 2), block 256 -> 2048 n-tiles total; K = 256 (8 WMMAs, dual chain)
__global__ __launch_bounds__(256)
void k_gemm_l1(const bf16_t* __restrict__ rb, const bf16_t* __restrict__ Pt,
               float* __restrict__ l1) {
  const int lane = threadIdx.x & 31, wave = threadIdx.x >> 5;
  const int half = lane >> 4, q = lane & 15;
  const int nt = blockIdx.x * 8 + wave, bt = blockIdx.y;
  const bf16_t* arow = rb + (size_t)(bt * 16 + q) * M_;
  const bf16_t* bcol = Pt + (size_t)(nt * 16 + q) * M_;
  v8f c0 = {}, c1 = {};
#pragma unroll
  for (int k0 = 0; k0 < M_; k0 += 64) {
    c0 = wmma_bf16(load_afrag(arow, k0, half), load_bfrag(bcol, k0, half), c0);
    c1 = wmma_bf16(load_afrag(arow, k0 + 32, half), load_bfrag(bcol, k0 + 32, half), c1);
  }
  float s[8];
#pragma unroll
  for (int v = 0; v < 8; ++v) s[v] = fabsf(c0[v] + c1[v]);
#pragma unroll
  for (int m = 1; m < 16; m <<= 1) {
#pragma unroll
    for (int v = 0; v < 8; ++v) s[v] += __shfl_xor(s[v], m, 32);
  }
  if (q == 0) {
#pragma unroll
    for (int v = 0; v < 8; ++v) atomicAdd(&l1[bt * 16 + v + 8 * half], s[v]);
  }
}

// ---------------- GEMM 4: u[b,n] = z[b,n] + sum_m r2[b,m]*Wt[n,m] ----------------
__global__ __launch_bounds__(256)
void k_gemm_u(const bf16_t* __restrict__ rb, const bf16_t* __restrict__ Wt,
              const float* __restrict__ z, float* __restrict__ u) {
  const int lane = threadIdx.x & 31, wave = threadIdx.x >> 5;
  const int half = lane >> 4, q = lane & 15;
  const int nt = blockIdx.x * 8 + wave, bt = blockIdx.y;
  const bf16_t* arow = rb + (size_t)(bt * 16 + q) * M_;
  const bf16_t* bcol = Wt + (size_t)(nt * 16 + q) * M_;
  v8f c0 = {}, c1 = {};
#pragma unroll
  for (int k0 = 0; k0 < M_; k0 += 64) {
    c0 = wmma_bf16(load_afrag(arow, k0, half), load_bfrag(bcol, k0, half), c0);
    c1 = wmma_bf16(load_afrag(arow, k0 + 32, half), load_bfrag(bcol, k0 + 32, half), c1);
  }
  const size_t col = (size_t)nt * 16 + q;
#pragma unroll
  for (int v = 0; v < 8; ++v) {
    size_t row = (size_t)(bt * 16 + v + 8 * half);
    u[row * N_ + col] = z[row * N_ + col] + (c0[v] + c1[v]);
  }
}

// ---------------- adaptive scalars (one wave; lane == batch row) ----------------
__global__ __launch_bounds__(32)
void k_scalars(const float* __restrict__ resl1, const float* __restrict__ yl1,
               const float* __restrict__ act, const float* __restrict__ c1,
               const float* __restrict__ c2, const float* __restrict__ c3,
               float progress, float* __restrict__ beta,
               float* __restrict__ thetaS, int* __restrict__ Kint) {
  int b = threadIdx.x;
  float rl = fmaxf(resl1[b], 1e-12f);
  float yl = fmaxf(yl1[b], 1e-12f);
  float rr = fminf(fmaxf(rl / yl, 0.0f), 1.0f);
  float th = fabsf(*c1) * rr;
  float sg = 1.0f / (1.0f + __expf(-(*c2)));
  beta[b] = sg * (act[b] / (float)N_);
  float ratio = fmaxf(yl / rl, 1.0f);
  float sig = 1.0f / (1.0f + __expf(-(fmaxf(__logf(ratio), 0.0f) - 1.0f)));
  float Kt = fabsf(*c3) * (float)N_ * fmaxf(sig, progress);
  Kt = fminf(fmaxf(Kt, 1.0f), 0.6f * (float)N_);
  float thsum = th, ksum = Kt;
  for (int m = 1; m < 32; m <<= 1) {
    thsum += __shfl_xor(thsum, m, 32);
    ksum += __shfl_xor(ksum, m, 32);
  }
  if (b == 0) {
    *thetaS = thsum * (1.0f / 32.0f);
    int K = (int)floorf(ksum * (1.0f / 32.0f));
    *Kint = K < 1 ? 1 : K;
  }
}

// ---------------- z = x + beta[b]*(x - x_prev) (momentum), + bf16 copy ----------------
__global__ __launch_bounds__(256)
void k_zupdate(const float* __restrict__ x, const float* __restrict__ xp,
               const float* __restrict__ beta, float* __restrict__ z,
               bf16_t* __restrict__ zb, int layer) {
  long i = (long)blockIdx.x * 256 + threadIdx.x;
  if (i < (long)B_ * N_) {
    int b = (int)(i / N_);
    float xi = x[i];
    float zi = (layer > 0) ? xi + beta[b] * (xi - xp[i]) : xi;
    z[i] = zi;
    zb[i] = (bf16_t)zi;
  }
}

// ---------------- tree scores: scores[i] = |u_i| + rho*scores[parent] ----------------
// one workgroup per batch row; binary-heap levels processed root->leaf
__global__ __launch_bounds__(1024)
void k_tree_scores(const float* __restrict__ u, float* __restrict__ scores) {
  int b = blockIdx.x;
  const float* ub = u + (size_t)b * N_;
  float* sb = scores + (size_t)b * N_;
  for (int i = threadIdx.x; i < N_; i += 1024) sb[i] = fabsf(ub[i]);
  __syncthreads();
  for (int d = 1; d <= 15; ++d) {
    int s = (1 << d) - 1;
    int e = (2 << d) - 1; if (e > N_) e = N_;
    for (int i = s + threadIdx.x; i < e; i += 1024)
      sb[i] = fabsf(ub[i]) + RHO_ * sb[(i - 1) >> 1];
    __syncthreads();
  }
}

// ---------------- exact K-th largest per row via MSB-first radix select ----------------
// scores >= 0 so uint bits are order-preserving
__global__ __launch_bounds__(256)
void k_radix_select(const float* __restrict__ scores, const int* __restrict__ Kint,
                    float* __restrict__ kth) {
  __shared__ unsigned hist[256];
  __shared__ unsigned sPrefix, sWant;
  int b = blockIdx.x;
  const float* sb = scores + (size_t)b * N_;
  if (threadIdx.x == 0) {
    sPrefix = 0u;
    int K = *Kint;
    sWant = (unsigned)(K < 1 ? 1 : K);
  }
  __syncthreads();
  for (int shift = 24; shift >= 0; shift -= 8) {
    hist[threadIdx.x] = 0u;
    __syncthreads();
    unsigned hiMask = (shift == 24) ? 0u : (0xFFFFFFFFu << (shift + 8));
    unsigned prefix = sPrefix;
    for (int i = threadIdx.x; i < N_; i += 256) {
      unsigned key = __float_as_uint(sb[i]);
      if ((key & hiMask) == prefix) atomicAdd(&hist[(key >> shift) & 255u], 1u);
    }
    __syncthreads();
    if (threadIdx.x == 0) {
      unsigned want = sWant, cum = 0u; int bin = 0;
      for (int j = 255; j >= 0; --j) {
        unsigned h = hist[j];
        if (cum + h >= want) { bin = j; sWant = want - cum; break; }
        cum += h;
      }
      sPrefix = prefix | ((unsigned)bin << shift);
    }
    __syncthreads();
  }
  if (threadIdx.x == 0) kth[b] = __uint_as_float(sPrefix);
}

// ---- mask + ancestor closure (LDS bitset) + soft shrink + fused outputs ----
// Writes x_new (f32), x_new (bf16, consumed by next layer's GEMM1), and the
// nonzero count of x_new (== next layer's `active`).
__global__ __launch_bounds__(1024)
void k_mask_shrink(const float* __restrict__ u, const float* __restrict__ scores,
                   const float* __restrict__ kth, const float* __restrict__ thetaS,
                   float* __restrict__ xnew, bf16_t* __restrict__ xnbf,
                   float* __restrict__ act) {
  __shared__ unsigned bits[N_ / 32];  // 4 KB bitset
  __shared__ int sred[1024];
  int b = blockIdx.x;
  float kv = kth[b], th = *thetaS;
  const float* sb = scores + (size_t)b * N_;
  const float* ub = u + (size_t)b * N_;
  for (int w = threadIdx.x; w < N_ / 32; w += 1024) bits[w] = 0u;
  __syncthreads();
  for (int i = threadIdx.x; i < N_; i += 1024)
    if (sb[i] >= kv) atomicOr(&bits[i >> 5], 1u << (i & 31));
  __syncthreads();
  for (int d = 15; d >= 1; --d) {            // leaf -> root closure
    int s = (1 << d) - 1;
    int e = (2 << d) - 1; if (e > N_) e = N_;
    for (int i = s + threadIdx.x; i < e; i += 1024) {
      if ((bits[i >> 5] >> (i & 31)) & 1u) {
        int p = (i - 1) >> 1;
        atomicOr(&bits[p >> 5], 1u << (p & 31));
      }
    }
    __syncthreads();
  }
  float* xb = xnew + (size_t)b * N_;
  bf16_t* xbb = xnbf + (size_t)b * N_;
  int cnt = 0;
  for (int i = threadIdx.x; i < N_; i += 1024) {
    float ui = ub[i];
    float a = fabsf(ui) - th;
    float v = (a > 0.0f) ? copysignf(a, ui) : 0.0f;
    float o = (((bits[i >> 5] >> (i & 31)) & 1u) ? v : 0.0f);
    xb[i] = o;
    xbb[i] = (bf16_t)o;
    cnt += (fabsf(o) > 1e-6f);
  }
  sred[threadIdx.x] = cnt;
  __syncthreads();
  for (int s = 512; s > 0; s >>= 1) {
    if (threadIdx.x < s) sred[threadIdx.x] += sred[threadIdx.x + s];
    __syncthreads();
  }
  if (threadIdx.x == 0) act[b] = (float)sred[0];
}

// ---------------- host orchestration ----------------
extern "C" void kernel_launch(void* const* d_in, const int* in_sizes, int n_in,
                              void* d_out, int out_size, void* d_ws, size_t ws_size,
                              hipStream_t stream) {
  (void)in_sizes; (void)n_in; (void)out_size; (void)ws_size;
  const float* y     = (const float*)d_in[0];
  const float* A     = (const float*)d_in[1];
  const float* W     = (const float*)d_in[2];
  const float* Apinv = (const float*)d_in[3];
  const float* c1    = (const float*)d_in[4];
  const float* c2    = (const float*)d_in[5];
  const float* c3    = (const float*)d_in[6];
  // d_in[7] (parent) / d_in[8] (depth) encode a binary heap; used analytically.

  // ---- workspace carve-out (~76 MiB) ----
  char* w = (char*)d_ws;
  size_t off = 0;
  auto take = [&](size_t bytes) -> char* {
    char* p = w + off;
    off = (off + bytes + 255) & ~(size_t)255;
    return p;
  };
  bf16_t* Abf   = (bf16_t*)take((size_t)M_ * N_ * 2);
  bf16_t* Aptbf = (bf16_t*)take((size_t)N_ * M_ * 2);
  bf16_t* Wtbf  = (bf16_t*)take((size_t)N_ * M_ * 2);
  float* xbuf[3];
  for (int i = 0; i < 3; ++i) xbuf[i] = (float*)take((size_t)B_ * N_ * 4);
  float*  z      = (float*)take((size_t)B_ * N_ * 4);
  float*  u      = (float*)take((size_t)B_ * N_ * 4);
  float*  scores = (float*)take((size_t)B_ * N_ * 4);
  bf16_t* xbf    = (bf16_t*)take((size_t)B_ * N_ * 2);  // bf16(x), maintained by mask_shrink
  bf16_t* zbf    = (bf16_t*)take((size_t)B_ * N_ * 2);  // bf16(z)
  bf16_t* ybf    = (bf16_t*)take((size_t)B_ * M_ * 2);
  bf16_t* rbf    = (bf16_t*)take((size_t)B_ * M_ * 2);
  float*  racc   = (float*)take((size_t)B_ * M_ * 4);
  float*  resl1  = (float*)take(B_ * 4);
  float*  yl1    = (float*)take(B_ * 4);
  float*  act    = (float*)take(B_ * 4);
  float*  beta   = (float*)take(B_ * 4);
  float*  kth    = (float*)take(B_ * 4);
  float*  thetaS = (float*)take(4);
  int*    Kint   = (int*)take(4);

  const long BN = (long)B_ * N_;
  const long AE = (long)M_ * N_;
  const dim3 gTiles(256, 2);          // n-tile GEMMs: 4096 waves
  const dim3 gXAT(16, 2, 8);          // K-split GEMM: 2048 waves

  // ---- one-time setup ----
  k_zero_f32<<<(BN + 255) / 256, 256, 0, stream>>>(xbuf[0], BN);       // x = 0
  k_zero_f32<<<(BN + 255) / 256, 256, 0, stream>>>(xbuf[1], BN);       // x_prev = 0
  k_zero_f32<<<(BN / 2 + 255) / 256, 256, 0, stream>>>((float*)xbf, BN / 2); // bf16(x)=0
  k_zero_f32<<<1, 256, 0, stream>>>(act, B_);                          // active(x0)=0
  k_cvt_bf16<<<(AE + 255) / 256, 256, 0, stream>>>(A, Abf, AE);
  k_cvt_bf16<<<(AE + 255) / 256, 256, 0, stream>>>(Apinv, Aptbf, AE);
  k_transpose_cvt<<<dim3(N_ / 32, M_ / 32), dim3(32, 8), 0, stream>>>(W, Wtbf);
  k_cvt_bf16<<<(B_ * M_ + 255) / 256, 256, 0, stream>>>(y, ybf, (long)B_ * M_);
  k_zero_f32<<<1, 256, 0, stream>>>(yl1, B_);
  k_gemm_l1<<<gTiles, 256, 0, stream>>>(ybf, Aptbf, yl1);  // y_l1

  float* xc = xbuf[0];  // current x
  float* xp = xbuf[1];  // x_prev
  float* xn = xbuf[2];  // x_new
  for (int k = 0; k < LAYERS_; ++k) {
    // residual = y - x A^T ; res_l1 = || |residual Apinv^T| ||_1 per row
    k_zero_f32<<<(B_ * M_ + 255) / 256, 256, 0, stream>>>(racc, (long)B_ * M_);
    k_gemm_xAT<<<gXAT, 256, 0, stream>>>(xbf, Abf, racc);
    k_residual<<<(B_ * M_ + 255) / 256, 256, 0, stream>>>(y, racc, rbf);
    k_zero_f32<<<1, 256, 0, stream>>>(resl1, B_);
    k_gemm_l1<<<gTiles, 256, 0, stream>>>(rbf, Aptbf, resl1);
    k_scalars<<<1, 32, 0, stream>>>(resl1, yl1, act, c1, c2, c3,
                                    (float)(k + 1) / (float)LAYERS_,
                                    beta, thetaS, Kint);
    // z = x + beta (x - x_prev); u = z + (y - z A^T) W
    k_zupdate<<<(BN + 255) / 256, 256, 0, stream>>>(xc, xp, beta, z, zbf, k);
    k_zero_f32<<<(B_ * M_ + 255) / 256, 256, 0, stream>>>(racc, (long)B_ * M_);
    k_gemm_xAT<<<gXAT, 256, 0, stream>>>(zbf, Abf, racc);
    k_residual<<<(B_ * M_ + 255) / 256, 256, 0, stream>>>(y, racc, rbf);
    k_gemm_u<<<gTiles, 256, 0, stream>>>(rbf, Wtbf, z, u);
    // tree operator
    k_tree_scores<<<B_, 1024, 0, stream>>>(u, scores);
    k_radix_select<<<B_, 256, 0, stream>>>(scores, Kint, kth);
    k_mask_shrink<<<B_, 1024, 0, stream>>>(u, scores, kth, thetaS, xn, xbf, act);
    // rotate (x_prev, x) <- (x, x_new)
    float* t = xp; xp = xc; xc = xn; xn = t;
  }

  hipMemcpyAsync(d_out, xc, (size_t)B_ * N_ * 4, hipMemcpyDeviceToDevice, stream);
}
